// ResBlock_50740743634945
// MI455X (gfx1250) — matrix operand order repaired and checked by reading, not compiled, not run on hardware
//
#include <hip/hip_runtime.h>
#include <hip/hip_bf16.h>
#include <math.h>

// ---- types ----------------------------------------------------------------
typedef __bf16 v8bf  __attribute__((ext_vector_type(8)));
typedef __bf16 v16bf __attribute__((ext_vector_type(16)));
typedef float  v8f   __attribute__((ext_vector_type(8)));
typedef float  f4v   __attribute__((ext_vector_type(4)));
typedef unsigned int v4u __attribute__((ext_vector_type(4)));
typedef int    v8i   __attribute__((ext_vector_type(8)));
typedef int    v4i   __attribute__((ext_vector_type(4)));

#define DIM   512
#define NROWS 65536
#define RS    520      // LDS row stride (elements): 512 + 8 -> +4 banks per row

#if defined(__has_builtin)
#  if __has_builtin(__builtin_amdgcn_tensor_load_to_lds) && __has_builtin(__builtin_amdgcn_s_wait_tensorcnt)
#    define HAVE_TDM 1
#  else
#    define HAVE_TDM 0
#  endif
#else
#  define HAVE_TDM 0
#endif

union FragBF { v16bf v; v8bf h[2]; };

__device__ __forceinline__ __bf16 f2bf(float f) { return (__bf16)f; }

// ---- kernel 1: W[k][n] f32 -> Wt[n][k] bf16 -------------------------------
__global__ __launch_bounds__(256) void transpose_w_bf16(
    const float* __restrict__ W, __bf16* __restrict__ Wt) {
  int idx = blockIdx.x * 256 + threadIdx.x;      // 0 .. 512*512-1
  int n = idx >> 9;
  int k = idx & 511;
  Wt[(size_t)n * DIM + k] = f2bf(W[(size_t)k * DIM + n]);
}

// ---- kernel 2: LayerNorm row -> bf16 --------------------------------------
__global__ __launch_bounds__(256) void ln_to_bf16(
    const float* __restrict__ x,
    const float* __restrict__ gamma,
    const float* __restrict__ beta,
    __bf16* __restrict__ hn) {
  const int wave = threadIdx.x >> 5;
  const int lane = threadIdx.x & 31;
  const int row  = blockIdx.x * 8 + wave;
  const float* xr = x + (size_t)row * DIM + lane * 16;

  f4v v[4];
  float s = 0.f, s2 = 0.f;
#pragma unroll
  for (int i = 0; i < 4; ++i) {
    v[i] = *(const f4v*)(xr + i * 4);
#pragma unroll
    for (int j = 0; j < 4; ++j) { s += v[i][j]; s2 += v[i][j] * v[i][j]; }
  }
#pragma unroll
  for (int m = 16; m >= 1; m >>= 1) {
    s  += __shfl_xor(s,  m, 32);
    s2 += __shfl_xor(s2, m, 32);
  }
  const float mu   = s * (1.f / DIM);
  const float var  = s2 * (1.f / DIM) - mu * mu;
  const float rstd = __frsqrt_rn(var + 1e-5f);

  const float* gp = gamma + lane * 16;
  const float* bp = beta  + lane * 16;
  v8bf o[2];
#pragma unroll
  for (int i = 0; i < 16; ++i) {
    float y = (v[i >> 2][i & 3] - mu) * rstd * gp[i] + bp[i];
    o[i >> 3][i & 7] = f2bf(y);
  }
  __bf16* hr = hn + (size_t)row * DIM + lane * 16;
  *(v8bf*)(hr)     = o[0];
  *(v8bf*)(hr + 8) = o[1];
}

// ---- TDM: one 2D descriptor loads the whole 64x512 bf16 B panel to LDS ----
// Tile: tile_dim0=512 elems (K, contiguous), tile_dim1=64 rows (N),
// tensor_dim0_stride=512, data_size=2B.  pad_enable: +4 DWORDs (16B) every
// 256 DWORDs (1 row) -> LDS row stride = 1040 B = RS elements.
#if HAVE_TDM
__device__ __forceinline__ void tdm_load_panel(unsigned int lds_addr,
                                               const __bf16* gsrc) {
  unsigned long long ga = (unsigned long long)(uintptr_t)gsrc;
  v4u g0;
  g0.x = 1u;                                         // count=1, user mode
  g0.y = lds_addr;                                   // lds_addr[31:0]
  g0.z = (unsigned int)ga;                           // global_addr[31:0]
  g0.w = (unsigned int)((ga >> 32) & 0x01FFFFFFu)    // global_addr[56:32]
         | (2u << 30);                               // type=2 ("image")

  const unsigned int d0 = (1u << 16)     // data_size = 2B
                        | (1u << 20)     // pad_enable
                        | (7u << 22)     // pad_interval: 256 DWORDs
                        | (3u << 25);    // pad_amount : 4 DWORDs
  const unsigned int tensor_d0 = 512, tensor_d1 = 64;
  const unsigned int tile_d0 = 512, tile_d1 = 64;
  const unsigned long long stride0 = 512;            // dim0 stride (elems)
  const unsigned int d1 = (tensor_d0 & 0xFFFFu) << 16;
  const unsigned int d2 = (tensor_d0 >> 16) | ((tensor_d1 & 0xFFFFu) << 16);
  const unsigned int d3 = (tensor_d1 >> 16) | ((tile_d0 & 0xFFFFu) << 16);
  const unsigned int d4 = (tile_d1 & 0xFFFFu);       // tile_dim2 = 0
  const unsigned int d5 = (unsigned int)stride0;
  const unsigned int d6 = (unsigned int)(stride0 >> 32);  // dim1 stride = 0
  v8i g1 = { (int)d0, (int)d1, (int)d2, (int)d3,
             (int)d4, (int)d5, (int)d6, 0 };
  v4i z4 = { 0, 0, 0, 0 };
  v8i z8 = { 0, 0, 0, 0, 0, 0, 0, 0 };
  // 6-arg form (clang-23 / therock-10.0 headers):
  // (uint32x4 g0, int32x8 g1, int32x4 g2, int32x4 g3, int32x8, i32 cpol)
  __builtin_amdgcn_tensor_load_to_lds(g0, g1, z4, z4, z8, 0);
}
#endif

// ---- kernel 3/4: bf16 WMMA GEMM, B panel staged in LDS --------------------
// block = 256 threads = 8 waves; wave tile 32x64 (8x v_wmma 16x16x32_bf16);
// block tile 256x64; grid = (M/256, 512/64).
template <bool GELU_EPI>
__global__ __launch_bounds__(256) void gemm_bf16(
    const __bf16* __restrict__ A,     // [M x 512] bf16
    const __bf16* __restrict__ Bt,    // [512 x 512] bf16, [n][k]
    const float*  __restrict__ bias,  // [512]
    const float*  __restrict__ resid, // x (residual path only)
    __bf16*       __restrict__ outBf, // g (gelu path only)
    float*        __restrict__ outF)  // out (residual path only)
{
  __shared__ __align__(128) __bf16 bpan[64 * RS];    // 66,560 B

  const int lane    = threadIdx.x & 31;
  const int wave    = threadIdx.x >> 5;
  const int group   = lane >> 4;      // 0: lanes 0-15, 1: lanes 16-31
  const int l16     = lane & 15;
  const int rowBase = blockIdx.x * 256 + wave * 32;
  const int colBase = blockIdx.y * 64;

  // ---- stage B panel (64 rows of Bt, full K) into LDS ----
#if HAVE_TDM
  if (threadIdx.x < 32) {            // wave 0 issues one TDM descriptor
    tdm_load_panel((unsigned int)(uintptr_t)&bpan[0],
                   Bt + ((size_t)colBase << 9));
    __builtin_amdgcn_s_wait_tensorcnt(0);
  }
#else
  for (int c = threadIdx.x; c < 4096; c += 256) {    // 4096 x 16B chunks
    int r = c >> 6, cc = c & 63;
    *(v8bf*)&bpan[r * RS + cc * 8] =
        *(const v8bf*)(Bt + (((size_t)(colBase + r)) << 9) + cc * 8);
  }
#endif
  __syncthreads();

  v8f acc[2][4] = {};
  const __bf16* aRow0 = A + (size_t)(rowBase + l16) * DIM;
  const __bf16* aRow1 = A + (size_t)(rowBase + 16 + l16) * DIM;

  // prologue: load A fragments for kb=0
  FragBF aCur[2], aNxt[2];
  aCur[0].h[0] = *(const v8bf*)(aRow0 + group * 8);
  aCur[0].h[1] = *(const v8bf*)(aRow0 + 16 + group * 8);
  aCur[1].h[0] = *(const v8bf*)(aRow1 + group * 8);
  aCur[1].h[1] = *(const v8bf*)(aRow1 + 16 + group * 8);

#pragma unroll 2
  for (int kb = 0; kb < DIM; kb += 32) {
    // software pipeline: fetch next A fragments before this step's WMMAs
    if (kb + 32 < DIM) {
      const int kn = kb + 32;
      aNxt[0].h[0] = *(const v8bf*)(aRow0 + kn + group * 8);
      aNxt[0].h[1] = *(const v8bf*)(aRow0 + kn + 16 + group * 8);
      aNxt[1].h[0] = *(const v8bf*)(aRow1 + kn + group * 8);
      aNxt[1].h[1] = *(const v8bf*)(aRow1 + kn + 16 + group * 8);
    }
#pragma unroll
    for (int t = 0; t < 4; ++t) {
      // B fragment from LDS: lane holds col N=l16 (+t*16), K=kb+group*8..+7 and kb+16+group*8..+7
      const __bf16* bp = &bpan[(t * 16 + l16) * RS + kb + group * 8];
      FragBF b;
      b.h[0] = *(const v8bf*)(bp);
      b.h[1] = *(const v8bf*)(bp + 16);
      acc[0][t] = __builtin_amdgcn_wmma_f32_16x16x32_bf16(
          false, aCur[0].v, false, b.v, (short)0, acc[0][t], false, false);
      acc[1][t] = __builtin_amdgcn_wmma_f32_16x16x32_bf16(
          false, aCur[1].v, false, b.v, (short)0, acc[1][t], false, false);
    }
    aCur[0] = aNxt[0];
    aCur[1] = aNxt[1];
  }

  // epilogue: C element (m, n): m = r + 8*group, n = l16 (per ISA C/D layout)
#pragma unroll
  for (int mt = 0; mt < 2; ++mt) {
#pragma unroll
    for (int t = 0; t < 4; ++t) {
      const int col = colBase + t * 16 + l16;
      const float bv = bias[col];
#pragma unroll
      for (int r = 0; r < 8; ++r) {
        const int row = rowBase + mt * 16 + r + group * 8;
        const size_t idx = (size_t)row * DIM + col;
        float vv = acc[mt][t][r] + bv;
        if (GELU_EPI) {
          float gel = 0.5f * vv * (1.0f + erff(vv * 0.70710678118654752f));
          outBf[idx] = f2bf(gel);
        } else {
          outF[idx] = resid[idx] + vv;
        }
      }
    }
  }
}

// ---- launch ---------------------------------------------------------------
extern "C" void kernel_launch(void* const* d_in, const int* in_sizes, int n_in,
                              void* d_out, int out_size, void* d_ws, size_t ws_size,
                              hipStream_t stream) {
  const float* x     = (const float*)d_in[0];
  const float* gamma = (const float*)d_in[1];
  const float* beta  = (const float*)d_in[2];
  const float* W1    = (const float*)d_in[3];
  const float* b1    = (const float*)d_in[4];
  const float* W2    = (const float*)d_in[5];
  const float* b2    = (const float*)d_in[6];
  float* out = (float*)d_out;

  char* ws = (char*)d_ws;
  __bf16* W1t = (__bf16*)ws;
  __bf16* W2t = (__bf16*)(ws + (size_t)DIM * DIM * 2);
  __bf16* hn  = (__bf16*)(ws + (size_t)2 * DIM * DIM * 2);
  __bf16* g   = hn + (size_t)NROWS * DIM;

  transpose_w_bf16<<<(DIM * DIM) / 256, 256, 0, stream>>>(W1, W1t);
  transpose_w_bf16<<<(DIM * DIM) / 256, 256, 0, stream>>>(W2, W2t);
  ln_to_bf16<<<NROWS / 8, 256, 0, stream>>>(x, gamma, beta, hn);

  dim3 grid(NROWS / 256, DIM / 64);
  gemm_bf16<true ><<<grid, 256, 0, stream>>>(hn, W1t, b1, nullptr, g, nullptr);
  gemm_bf16<false><<<grid, 256, 0, stream>>>(g,  W2t, b2, x, nullptr, out);
}
